// MultiheadAttentionWithRoPE_44263932953336
// MI455X (gfx1250) — compile-verified, matrix operand order checked by reading, not run on hardware
//
#include <hip/hip_runtime.h>
#include <math.h>

// ---------------------------------------------------------------------------
// MHA + RoPE forward for MI455X (gfx1250, wave32, WMMA + async-to-LDS).
// B=2 T=2048 C=1024 H=16 HD=64.  ~69 GFLOP vs ~100MB traffic -> compute bound.
// All matmuls via v_wmma_f32_16x16x32_bf16; all tile staging via
// global_load_async_to_lds_b128 (ASYNCcnt) with double-buffered LDS.
// ---------------------------------------------------------------------------

typedef __attribute__((ext_vector_type(16))) __bf16 v16bf;
typedef __attribute__((ext_vector_type(8)))  __bf16 v8bf;
typedef __attribute__((ext_vector_type(4)))  __bf16 v4bf;
typedef __attribute__((ext_vector_type(8)))  float  v8f;
typedef __attribute__((ext_vector_type(4)))  float  v4f;

static constexpr int Bb  = 2;
static constexpr int Ts  = 2048;
static constexpr int Cc  = 1024;
static constexpr int Hh  = 16;
static constexpr int HDd = 64;
static constexpr int GM  = Bb * Ts;   // 4096
static constexpr int GN  = Cc;        // 1024
static constexpr int GK  = Cc;        // 1024

__device__ __forceinline__ v8f wmma_bf16(v16bf a, v16bf b, v8f c) {
  return __builtin_amdgcn_wmma_f32_16x16x32_bf16(false, a, false, b,
                                                 (short)0, c, false, false);
}

// CDNA5 async copy: 16B global -> LDS, tracked by ASYNCcnt.
// Low 32 bits of a generic pointer to LDS are the wave-relative LDS address.
__device__ __forceinline__ void async_ld_b128(void* lds, const void* gptr) {
  unsigned loff = (unsigned)(uintptr_t)lds;
  asm volatile("global_load_async_to_lds_b128 %0, %1, off"
               :: "v"(loff), "v"((unsigned long long)(uintptr_t)gptr)
               : "memory");
}
__device__ __forceinline__ void wait_async0() {
  asm volatile("s_wait_asynccnt 0x0" ::: "memory");
}

// 16-lane-striped bf16 A/B fragment per the CDNA5 ISA layout:
// lane l: row/col = l%16, K-base = (l/16)*8; halves [base..base+7] ->
// VGPR0..3, [base+16..base+23] -> VGPR4..7.
__device__ __forceinline__ v16bf make_frag(const __bf16* p0, const __bf16* p1) {
  v8bf lo = *(const v8bf*)p0;
  v8bf hi = *(const v8bf*)p1;
  return __builtin_shufflevector(lo, hi, 0, 1, 2, 3, 4, 5, 6, 7,
                                         8, 9, 10, 11, 12, 13, 14, 15);
}

// ---------------------------------------------------------------------------
// f32 -> bf16 repack (one-time; removes conversion from GEMM hot loops)
// ---------------------------------------------------------------------------
__global__ __launch_bounds__(256) void pack_bf16_kernel(
    const float* __restrict__ in, __bf16* __restrict__ out, int n) {
  const int i = (blockIdx.x * 256 + threadIdx.x) * 4;
  if (i < n) {
    v4f f = *(const v4f*)(in + i);
    v4bf o = {(__bf16)f.x, (__bf16)f.y, (__bf16)f.z, (__bf16)f.w};
    *(v4bf*)(out + i) = o;
  }
}

// ---------------------------------------------------------------------------
// GEMM:  C[M,N] = A[M,K] @ W[N,K]^T   (bf16 in, f32 out)
// 256 threads = 8 waves; 128x128x32 tile; wave -> 32x64 (2x4 fragments).
// Double-buffered LDS fed by async-to-LDS; 1 barrier per k-step.
// ---------------------------------------------------------------------------
__device__ __forceinline__ void gemm_issue(const __bf16* A, const __bf16* W,
                                           __bf16* asb, __bf16* bsb,
                                           int tid, int m0, int n0, int k0) {
#pragma unroll
  for (int o = 0; o < 2; ++o) {
    const int s   = tid * 2 + o;        // 512 16B slots per matrix
    const int row = s >> 2;             // 0..127
    const int sg  = (s & 3) * 8;        // 0,8,16,24 bf16
    async_ld_b128(&asb[row * 32 + sg], A + (size_t)(m0 + row) * GK + k0 + sg);
    async_ld_b128(&bsb[row * 32 + sg], W + (size_t)(n0 + row) * GK + k0 + sg);
  }
}

__global__ __launch_bounds__(256) void gemm_nt_kernel(
    const __bf16* __restrict__ A, const __bf16* __restrict__ W,
    float* __restrict__ C) {
  __shared__ __attribute__((aligned(16))) __bf16 As[2][128 * 32];
  __shared__ __attribute__((aligned(16))) __bf16 Bs[2][128 * 32];

  const int tid  = threadIdx.x;
  const int lane = tid & 31;
  const int wid  = tid >> 5;
  const int wM   = wid & 3;   // 4 waves along M (32 rows each)
  const int wN   = wid >> 2;  // 2 waves along N (64 cols each)
  const int m0   = blockIdx.y * 128;
  const int n0   = blockIdx.x * 128;
  const int r    = lane & 15;
  const int hwv  = lane >> 4;
  const int base = hwv * 8;

  const v8f zero = {0.f, 0.f, 0.f, 0.f, 0.f, 0.f, 0.f, 0.f};
  v8f acc[2][4];
#pragma unroll
  for (int i = 0; i < 2; ++i)
#pragma unroll
    for (int j = 0; j < 4; ++j) acc[i][j] = zero;

  gemm_issue(A, W, As[0], Bs[0], tid, m0, n0, 0);

  int buf = 0;
  for (int k0 = 0; k0 < GK; k0 += 32, buf ^= 1) {
    wait_async0();
    __syncthreads();
    if (k0 + 32 < GK)
      gemm_issue(A, W, As[buf ^ 1], Bs[buf ^ 1], tid, m0, n0, k0 + 32);

    v16bf af[2], bfr[4];
#pragma unroll
    for (int i = 0; i < 2; ++i) {
      const __bf16* p = &As[buf][(wM * 32 + i * 16 + r) * 32];
      af[i] = make_frag(p + base, p + base + 16);
    }
#pragma unroll
    for (int j = 0; j < 4; ++j) {
      const __bf16* p = &Bs[buf][(wN * 64 + j * 16 + r) * 32];
      bfr[j] = make_frag(p + base, p + base + 16);
    }
#pragma unroll
    for (int i = 0; i < 2; ++i)
#pragma unroll
      for (int j = 0; j < 4; ++j)
        acc[i][j] = wmma_bf16(af[i], bfr[j], acc[i][j]);
  }

#pragma unroll
  for (int i = 0; i < 2; ++i)
#pragma unroll
    for (int j = 0; j < 4; ++j)
#pragma unroll
      for (int v = 0; v < 8; ++v) {
        const int gm = m0 + wM * 32 + i * 16 + hwv * 8 + v;
        const int gn = n0 + wN * 64 + j * 16 + r;
        C[(size_t)gm * GN + gn] = acc[i][j][v];
      }
}

// ---------------------------------------------------------------------------
// RoPE + pack: f32 [B,T,C] projections -> bf16 Q,K [B,H,T,64], V^T [B,H,64,T]
// ---------------------------------------------------------------------------
__global__ __launch_bounds__(256) void rope_pack_kernel(
    const float* __restrict__ pq, const float* __restrict__ pk,
    const float* __restrict__ pv, const float* __restrict__ freqs,
    __bf16* __restrict__ qb, __bf16* __restrict__ kb, __bf16* __restrict__ vt) {
  const int idx = blockIdx.x * 256 + threadIdx.x;   // 2^21 threads
  const int p = idx & 31;
  const int t = (idx >> 5) & (Ts - 1);
  const int h = (idx >> 16) & (Hh - 1);
  const int b = idx >> 20;

  const size_t src = ((size_t)(b * Ts + t)) * Cc + h * HDd + 2 * p;
  const float f = freqs[t * (HDd / 2) + p];
  float cs, sn;
  __sincosf(f, &sn, &cs);

  const float q0 = pq[src], q1 = pq[src + 1];
  const float k0 = pk[src], k1 = pk[src + 1];
  const float v0 = pv[src], v1 = pv[src + 1];

  const size_t dst = ((size_t)((b * Hh + h) * Ts + t)) * HDd + 2 * p;
  qb[dst]     = (__bf16)(q0 * cs - q1 * sn);
  qb[dst + 1] = (__bf16)(q0 * sn + q1 * cs);
  kb[dst]     = (__bf16)(k0 * cs - k1 * sn);
  kb[dst + 1] = (__bf16)(k0 * sn + k1 * cs);

  const size_t vd = ((size_t)((b * Hh + h) * HDd + 2 * p)) * Ts + t;
  vt[vd]      = (__bf16)v0;
  vt[vd + Ts] = (__bf16)v1;
}

// ---------------------------------------------------------------------------
// Flash attention: 1 block = 4 waves = 64 query rows of one (b,h).
// K/V tiles (shared by all 4 waves) staged once per 32-key step through
// double-buffered LDS via async-to-LDS; softmax via per-wave LDS regions
// (same-wave LDS is in-order -> no extra barriers).
// ---------------------------------------------------------------------------
__device__ __forceinline__ void attn_issue(const __bf16* Kh, const __bf16* Vh,
                                           __bf16* ks, __bf16* vs,
                                           int tid, int k0) {
#pragma unroll
  for (int o = 0; o < 2; ++o) {
    const int s = tid * 2 + o;                 // 256 16B slots per matrix
    const int kr = s >> 3, ksg = (s & 7) * 8;  // K tile: 32 keys x 64 hd
    async_ld_b128(&ks[kr * 64 + ksg], Kh + (size_t)(k0 + kr) * HDd + ksg);
    const int vr = s >> 2, vsg = (s & 3) * 8;  // V tile: 64 hd x 32 keys (V^T)
    async_ld_b128(&vs[vr * 32 + vsg], Vh + (size_t)vr * Ts + k0 + vsg);
  }
}

__global__ __launch_bounds__(128) void attn_kernel(
    const __bf16* __restrict__ Qb, const __bf16* __restrict__ Kb,
    const __bf16* __restrict__ Vt, __bf16* __restrict__ y) {
  __shared__ __attribute__((aligned(16))) __bf16 Ks[2][32 * 64];
  __shared__ __attribute__((aligned(16))) __bf16 Vs[2][64 * 32];
  __shared__ __attribute__((aligned(16))) float  Sb[4][16][32];
  __shared__ __attribute__((aligned(16))) __bf16 Pb[4][16][32];
  __shared__ float mr[4][16], lr[4][16], ar[4][16], ir[4][16];

  const int tid  = threadIdx.x;
  const int lane = tid & 31;
  const int wave = tid >> 5;
  const int r    = lane & 15;
  const int hwv  = lane >> 4;
  const int base = hwv * 8;
  const int h = blockIdx.y, b = blockIdx.z;

  const size_t head = ((size_t)(b * Hh + h)) * Ts * HDd;
  const __bf16* Qh = Qb + head;
  const __bf16* Kh = Kb + head;
  const __bf16* Vh = Vt + head;   // [64][T] transposed layout, same size
  const int t0 = blockIdx.x * 64 + wave * 16;

  const __bf16* qp = Qh + (size_t)(t0 + r) * HDd;
  const v16bf q0 = make_frag(qp + base, qp + base + 16);
  const v16bf q1 = make_frag(qp + 32 + base, qp + 32 + base + 16);

  const v8f zero = {0.f, 0.f, 0.f, 0.f, 0.f, 0.f, 0.f, 0.f};
  v8f o[4];
#pragma unroll
  for (int c = 0; c < 4; ++c) o[c] = zero;

  if (lane < 16) { mr[wave][lane] = -1e30f; lr[wave][lane] = 0.f; }

  attn_issue(Kh, Vh, Ks[0], Vs[0], tid, 0);

  int buf = 0;
  for (int j = 0; j < Ts / 32; ++j, buf ^= 1) {
    wait_async0();
    __syncthreads();
    if (j + 1 < Ts / 32)
      attn_issue(Kh, Vh, Ks[buf ^ 1], Vs[buf ^ 1], tid, (j + 1) * 32);

    v8f s[2];
#pragma unroll
    for (int nc = 0; nc < 2; ++nc) {
      const __bf16* kp = &Ks[buf][(nc * 16 + r) * 64];
      const v16bf kf0 = make_frag(kp + base, kp + base + 16);
      const v16bf kf1 = make_frag(kp + 32 + base, kp + 32 + base + 16);
      v8f a = zero;
      a = wmma_bf16(q0, kf0, a);
      a = wmma_bf16(q1, kf1, a);
      s[nc] = a;
    }
#pragma unroll
    for (int nc = 0; nc < 2; ++nc)
#pragma unroll
      for (int v = 0; v < 8; ++v)
        Sb[wave][hwv * 8 + v][nc * 16 + r] = s[nc][v] * 0.125f;  // 1/sqrt(64)

    if (lane < 16) {  // one lane per query row: online softmax over 32 keys
      const float mo = mr[wave][lane];
      float mj = mo;
      for (int c2 = 0; c2 < 32; ++c2) mj = fmaxf(mj, Sb[wave][lane][c2]);
      const float al = __expf(mo - mj);
      float ls = 0.f;
      for (int c2 = 0; c2 < 32; ++c2) {
        const float pv = __expf(Sb[wave][lane][c2] - mj);
        ls += pv;
        Pb[wave][lane][c2] = (__bf16)pv;
      }
      mr[wave][lane] = mj;
      lr[wave][lane] = lr[wave][lane] * al + ls;
      ar[wave][lane] = al;
    }

#pragma unroll
    for (int v = 0; v < 8; ++v) {   // rescale O by alpha(row)
      const float al = ar[wave][hwv * 8 + v];
      o[0][v] *= al; o[1][v] *= al; o[2][v] *= al; o[3][v] *= al;
    }

    const __bf16* pp = &Pb[wave][r][0];
    const v16bf pf = make_frag(pp + base, pp + base + 16);
#pragma unroll
    for (int c = 0; c < 4; ++c) {
      const __bf16* vp = &Vs[buf][(c * 16 + r) * 32];
      const v16bf vf = make_frag(vp + base, vp + base + 16);
      o[c] = wmma_bf16(pf, vf, o[c]);
    }
  }

  if (lane < 16) ir[wave][lane] = 1.0f / lr[wave][lane];
  __bf16* yp = y + ((size_t)b * Ts + t0) * Cc + h * HDd;
#pragma unroll
  for (int c = 0; c < 4; ++c)
#pragma unroll
    for (int v = 0; v < 8; ++v)
      yp[(size_t)(hwv * 8 + v) * Cc + c * 16 + r] =
          (__bf16)(o[c][v] * ir[wave][hwv * 8 + v]);
}

// ---------------------------------------------------------------------------
extern "C" void kernel_launch(void* const* d_in, const int* in_sizes, int n_in,
                              void* d_out, int out_size, void* d_ws,
                              size_t ws_size, hipStream_t stream) {
  (void)in_sizes; (void)n_in; (void)out_size; (void)ws_size;
  const float* query = (const float*)d_in[0];
  const float* key_  = (const float*)d_in[1];
  const float* value = (const float*)d_in[2];
  const float* freqs = (const float*)d_in[3];
  const float* Wq    = (const float*)d_in[4];
  const float* Wk    = (const float*)d_in[5];
  const float* Wv    = (const float*)d_in[6];
  const float* Wo    = (const float*)d_in[7];
  float* out = (float*)d_out;

  char* ws = (char*)d_ws;
  const size_t MB     = 1u << 20;
  const size_t ACT_BF = (size_t)GM * GK * 2;  // 8 MB
  const size_t W_BF   = (size_t)GN * GK * 2;  // 2 MB
  const size_t PROJ   = (size_t)GM * GN * 4;  // 16 MB
  __bf16* qin  = (__bf16*)(ws);
  __bf16* kin  = (__bf16*)(ws + 8 * MB);
  __bf16* vin  = (__bf16*)(ws + 16 * MB);
  __bf16* wqb  = (__bf16*)(ws + 24 * MB);
  __bf16* wkb  = (__bf16*)(ws + 26 * MB);
  __bf16* wvb  = (__bf16*)(ws + 28 * MB);
  __bf16* wob  = (__bf16*)(ws + 30 * MB);
  float*  projQ = (float*)(ws + 32 * MB);
  float*  projK = (float*)(ws + 48 * MB);
  float*  projV = (float*)(ws + 64 * MB);
  __bf16* Qbf  = (__bf16*)(ws + 80 * MB);
  __bf16* Kbf  = (__bf16*)(ws + 88 * MB);
  __bf16* Vtr  = (__bf16*)(ws + 96 * MB);
  __bf16* ybf  = qin;  // qin is dead after the Q projection GEMM
  (void)ACT_BF; (void)W_BF; (void)PROJ;

  const int nact = GM * GK, nw = GN * GK;
  pack_bf16_kernel<<<nact / 1024, 256, 0, stream>>>(query, qin, nact);
  pack_bf16_kernel<<<nact / 1024, 256, 0, stream>>>(key_,  kin, nact);
  pack_bf16_kernel<<<nact / 1024, 256, 0, stream>>>(value, vin, nact);
  pack_bf16_kernel<<<nw / 1024, 256, 0, stream>>>(Wq, wqb, nw);
  pack_bf16_kernel<<<nw / 1024, 256, 0, stream>>>(Wk, wkb, nw);
  pack_bf16_kernel<<<nw / 1024, 256, 0, stream>>>(Wv, wvb, nw);
  pack_bf16_kernel<<<nw / 1024, 256, 0, stream>>>(Wo, wob, nw);

  const dim3 ggrid(GN / 128, GM / 128);  // (8, 32)
  gemm_nt_kernel<<<ggrid, 256, 0, stream>>>(qin, wqb, projQ);
  gemm_nt_kernel<<<ggrid, 256, 0, stream>>>(kin, wkb, projK);
  gemm_nt_kernel<<<ggrid, 256, 0, stream>>>(vin, wvb, projV);

  const int nrope = Bb * Hh * Ts * (HDd / 2);  // 2^21
  rope_pack_kernel<<<nrope / 256, 256, 0, stream>>>(projQ, projK, projV, freqs,
                                                    Qbf, Kbf, Vtr);

  attn_kernel<<<dim3(Ts / 64, Hh, Bb), 128, 0, stream>>>(Qbf, Kbf, Vtr, ybf);

  gemm_nt_kernel<<<ggrid, 256, 0, stream>>>(ybf, wob, out);
}
